// MGUQ_network_18159121728153
// MI455X (gfx1250) — compile-verified, matrix operand order checked by reading, not run on hardware
//
#include <hip/hip_runtime.h>
#include <hip/hip_bf16.h>

// ---------------- problem constants ----------------
#define BB   4096
#define NN   128
#define HH   256
#define IDD  64
#define RANKK 16
#define DISTF 36
#define IN1  384            // ID + ID + H
#define G3H  768            // 3*H

// d_out layout (floats): loss | predict_mean | mean | Cov | lower | upper
#define OFF_LOSS  0
#define OFF_PM    1
#define OFF_MEAN  (1 + BB)
#define OFF_COV   (OFF_MEAN + (size_t)BB*NN)
#define OFF_LOWER (OFF_COV + (size_t)BB*NN*NN)
#define OFF_UPPER (OFF_LOWER + BB)

typedef __attribute__((ext_vector_type(16))) __bf16 v16bf;
typedef __attribute__((ext_vector_type(8)))  float  v8f;

// ---- WMMA fragment load: 16-bit A (16x32) or B^T (row-major, lane=row/col) ----
// Per ISA layout: lane<16 -> K = {0..7, 16..23}+0 ; lane>=16 -> +8.
// p must point at element [ (lane&15) row of tile ][ k0 + kh ].
__device__ __forceinline__ v16bf ldfrag(const __bf16* p) {
  v16bf v;
#pragma unroll
  for (int i = 0; i < 8; ++i) v[i] = p[i];
#pragma unroll
  for (int i = 0; i < 8; ++i) v[8 + i] = p[16 + i];
  return v;
}

__device__ __forceinline__ v8f wmma_bf16(v16bf a, v16bf b, v8f c) {
  return __builtin_amdgcn_wmma_f32_16x16x32_bf16(false, a, false, b, (short)0, c,
                                                 false, false);
}

// ---- CDNA5 async global->LDS copy (ASYNCcnt path), 16B per lane ----
__device__ __forceinline__ void async_g2l_b128(unsigned lds_byte_addr,
                                               const void* gaddr) {
  asm volatile("global_load_async_to_lds_b128 %0, %1, off"
               :: "v"(lds_byte_addr), "v"(gaddr)
               : "memory");
}
__device__ __forceinline__ void wait_async0() {
  asm volatile("s_wait_asynccnt 0x0" ::: "memory");
}

// =====================================================================
// Kernel 0: convert/transpose weights to bf16 B^T form (row=n, contiguous k)
// =====================================================================
__global__ __launch_bounds__(256) void k_prep(
    const float* __restrict__ W1, const float* __restrict__ W2,
    const float* __restrict__ Wih, const float* __restrict__ Whh,
    __bf16* __restrict__ W1t, __bf16* __restrict__ W2t,
    __bf16* __restrict__ Wihb, __bf16* __restrict__ Whhb) {
  int i = blockIdx.x * 256 + threadIdx.x;
  if (i < HH * IN1) { int n = i / IN1, k = i % IN1; W1t[i] = (__bf16)W1[k * HH + n]; }
  if (i < HH * HH)  { int n = i / HH,  k = i % HH;  W2t[i] = (__bf16)W2[k * HH + n]; }
  if (i < G3H * HH) { Wihb[i] = (__bf16)Wih[i]; Whhb[i] = (__bf16)Whh[i]; }
}

// =====================================================================
// Kernel 1: features (embed + dist@Wd) -> MLP (two WMMA GEMMs) -> x (bf16)
// One block = 32 rows of the flattened (B*N) dimension. 8 waves.
// =====================================================================
__global__ __launch_bounds__(256) void k_featmlp(
    const int* __restrict__ xs, const float* __restrict__ dist,
    const int* __restrict__ start_ts,
    const float* __restrict__ seg_emb, const float* __restrict__ slice_emb,
    const float* __restrict__ Wd, const float* __restrict__ bd,
    const __bf16* __restrict__ W1t, const float* __restrict__ b1,
    const __bf16* __restrict__ W2t, const float* __restrict__ b2,
    __bf16* __restrict__ xout) {
  __shared__ __bf16 fbuf[32][IN1];
  __shared__ __bf16 h1[32][HH];

  const int row0 = blockIdx.x * 32;
  const int tid  = threadIdx.x;
  const int wave = tid >> 5, lane = tid & 31;
  const int lr16 = lane & 15, kh = (lane >> 4) * 8;

  // ---- stage features ----
  for (int idx = tid; idx < 32 * 128; idx += 256) {       // embeddings
    int lr = idx >> 7, c = idx & 127;
    int p = row0 + lr, b = p >> 7;                        // N = 128
    float v = (c < 64) ? seg_emb[(size_t)xs[p] * IDD + c]
                       : slice_emb[(size_t)start_ts[b] * IDD + (c - 64)];
    fbuf[lr][c] = (__bf16)v;
  }
  for (int idx = tid; idx < 32 * HH; idx += 256) {        // real = dist@Wd + bd
    int lr = idx >> 8, j = idx & 255;
    const float* df = dist + (size_t)(row0 + lr) * DISTF;
    float acc = bd[j];
#pragma unroll
    for (int k = 0; k < DISTF; ++k) acc += df[k] * Wd[k * HH + j];
    fbuf[lr][128 + j] = (__bf16)acc;
  }
  __syncthreads();

  // ---- GEMM1: h1 = relu(f @ W1 + b1) ; M=32(2 tiles) N=256(16) K=384 ----
#pragma unroll
  for (int i = 0; i < 4; ++i) {
    int t = wave * 4 + i, mt = t >> 4, nt = t & 15;
    v8f c; float bv = b1[nt * 16 + lr16];
#pragma unroll
    for (int r = 0; r < 8; ++r) c[r] = bv;
    const __bf16* ap = &fbuf[mt * 16 + lr16][kh];
    const __bf16* bp = W1t + (size_t)(nt * 16 + lr16) * IN1 + kh;
#pragma unroll
    for (int kc = 0; kc < IN1; kc += 32)
      c = wmma_bf16(ldfrag(ap + kc), ldfrag(bp + kc), c);
    int rb = mt * 16 + ((lane >> 4) << 3), col = nt * 16 + lr16;
#pragma unroll
    for (int r = 0; r < 8; ++r) {
      float v = c[r]; v = v > 0.f ? v : 0.f;
      h1[rb + r][col] = (__bf16)v;
    }
  }
  __syncthreads();

  // ---- GEMM2: x = h1 @ W2 + b2 ; K=256 ----
#pragma unroll
  for (int i = 0; i < 4; ++i) {
    int t = wave * 4 + i, mt = t >> 4, nt = t & 15;
    v8f c; float bv = b2[nt * 16 + lr16];
#pragma unroll
    for (int r = 0; r < 8; ++r) c[r] = bv;
    const __bf16* ap = &h1[mt * 16 + lr16][kh];
    const __bf16* bp = W2t + (size_t)(nt * 16 + lr16) * HH + kh;
#pragma unroll
    for (int kc = 0; kc < HH; kc += 32)
      c = wmma_bf16(ldfrag(ap + kc), ldfrag(bp + kc), c);
    int rb = mt * 16 + ((lane >> 4) << 3), col = nt * 16 + lr16;
#pragma unroll
    for (int r = 0; r < 8; ++r)
      xout[(size_t)(row0 + rb + r) * HH + col] = (__bf16)c[r];
  }
}

// =====================================================================
// Kernel 2: GRU. One block owns 32 batch rows for all 128 steps.
// x_t tiles double-buffered in LDS via async global->LDS (ASYNCcnt);
// per step xg = x_t@W_ih^T + b_ih ; hg = h@W_hh^T + b_hh (WMMA), gates.
// Dynamic LDS: xb0(16K) xb1(16K) hb(16K) xg(96K) hg(96K) = 240 KB.
// =====================================================================
__global__ __launch_bounds__(256) void k_gru(
    const __bf16* __restrict__ x, const __bf16* __restrict__ Wih,
    const __bf16* __restrict__ Whh, const float* __restrict__ b_ih,
    const float* __restrict__ b_hh, const int* __restrict__ lengths,
    float* __restrict__ out) {
  extern __shared__ char smem[];
  __bf16* xb0 = (__bf16*)smem;           // [32][256]
  __bf16* xb1 = xb0 + 32 * HH;           // [32][256]
  __bf16* hb  = xb1 + 32 * HH;           // [32][256]
  float*  xg  = (float*)(hb + 32 * HH);  // [32][768]
  float*  hg  = xg + 32 * G3H;           // [32][768]

  const int b0 = blockIdx.x * 32;
  const int tid = threadIdx.x, wave = tid >> 5, lane = tid & 31;
  const int lr16 = lane & 15, kh = (lane >> 4) * 8;
  const char* xbytes = (const char*)x;

  // issue async copy of the full 16KB x_t tile: 4 x 16B per thread
  auto issue_xtile = [&](int t, __bf16* dst) {
    unsigned ldsbase = (unsigned)(size_t)dst;   // low 32 bits = LDS byte offset
#pragma unroll
    for (int k2 = 0; k2 < 4; ++k2) {
      int jb = (k2 * 256 + tid) * 16;           // byte index in tile [0,16384)
      int lr = jb >> 9, cb = jb & 511;          // row (512B per row), col byte
      const char* g = xbytes + ((size_t)(b0 + lr) * NN + t) * (HH * 2) + cb;
      async_g2l_b128(ldsbase + (unsigned)jb, g);
    }
  };

  for (int i = tid; i < 32 * HH; i += 256) hb[i] = (__bf16)0.0f;
  issue_xtile(0, xb0);
  wait_async0();
  __syncthreads();

  __bf16* xcur = xb0;
  __bf16* xnxt = xb1;

  for (int t = 0; t < NN; ++t) {
    // prefetch next step's x tile while this step computes
    if (t + 1 < NN) issue_xtile(t + 1, xnxt);

    // 2 GEMMs x 2 Mtiles x 48 Ntiles = 192 tiles, 24 per wave, K=256
    for (int i = 0; i < 24; ++i) {
      int tt = wave * 24 + i;
      int g  = (tt >= 96) ? 1 : 0;
      int t2 = tt - g * 96;
      int mt = t2 / 48, nt = t2 % 48;
      const __bf16* A   = g ? hb  : xcur;
      const __bf16* Bt  = g ? Whh : Wih;
      const float*  bia = g ? b_hh : b_ih;
      float*        O   = g ? hg  : xg;
      v8f c; float bv = bia[nt * 16 + lr16];
#pragma unroll
      for (int r = 0; r < 8; ++r) c[r] = bv;
      const __bf16* ap = A + (mt * 16 + lr16) * HH + kh;
      const __bf16* bp = Bt + (size_t)(nt * 16 + lr16) * HH + kh;
#pragma unroll
      for (int kc = 0; kc < HH; kc += 32)
        c = wmma_bf16(ldfrag(ap + kc), ldfrag(bp + kc), c);
      int rb = mt * 16 + ((lane >> 4) << 3), col = nt * 16 + lr16;
#pragma unroll
      for (int r = 0; r < 8; ++r) O[(rb + r) * G3H + col] = c[r];
    }
    __syncthreads();

    // gate math (torch order r,z,n) + masked output store
    for (int i = tid; i < 32 * HH; i += 256) {
      int lr = i >> 8, j = i & 255;
      float xr = xg[lr * G3H + j],         hr = hg[lr * G3H + j];
      float xz = xg[lr * G3H + HH + j],    hz = hg[lr * G3H + HH + j];
      float xn = xg[lr * G3H + 2*HH + j],  hn = hg[lr * G3H + 2*HH + j];
      float r = 1.f / (1.f + __expf(-(xr + hr)));
      float z = 1.f / (1.f + __expf(-(xz + hz)));
      float n = tanhf(xn + r * hn);
      float hold = (float)hb[lr * HH + j];
      float hnew = (1.f - z) * n + z * hold;
      hb[lr * HH + j] = (__bf16)hnew;
      float m = (t < lengths[b0 + lr]) ? 1.f : 0.f;
      out[((size_t)(b0 + lr) * NN + t) * HH + j] = hnew * m;
    }
    // next tile must be fully in LDS before anyone reads it next iteration
    wait_async0();
    __syncthreads();

    __bf16* tmp = xcur; xcur = xnxt; xnxt = tmp;
  }
}

// =====================================================================
// Kernel 3: heads. One wave per (b,t) row: mean, Dm, Vm.
// =====================================================================
__global__ __launch_bounds__(256) void k_heads(
    const float* __restrict__ out, const float* __restrict__ Wm,
    const float* __restrict__ bm, const float* __restrict__ Wdg,
    const float* __restrict__ bdg, const float* __restrict__ Wlr,
    const float* __restrict__ blr, const int* __restrict__ lengths,
    float* __restrict__ mean_out, float* __restrict__ Vm,
    float* __restrict__ Dm) {
  int p = blockIdx.x * 8 + (threadIdx.x >> 5);
  int lane = threadIdx.x & 31;
  const float* o = out + (size_t)p * HH;
  float am = 0.f, ad = 0.f, av[RANKK];
#pragma unroll
  for (int r = 0; r < RANKK; ++r) av[r] = 0.f;
  for (int j = lane; j < HH; j += 32) {
    float v = o[j];
    am += v * Wm[j];
    ad += v * Wdg[j];
#pragma unroll
    for (int r = 0; r < RANKK; ++r) av[r] += v * Wlr[j * RANKK + r];
  }
#pragma unroll
  for (int s = 16; s; s >>= 1) {
    am += __shfl_xor(am, s, 32);
    ad += __shfl_xor(ad, s, 32);
#pragma unroll
    for (int r = 0; r < RANKK; ++r) av[r] += __shfl_xor(av[r], s, 32);
  }
  if (lane == 0) {
    int b = p >> 7, t = p & 127;
    float m = (t < lengths[b]) ? 1.f : 0.f;
    mean_out[p] = am + bm[0];
    float rd = ad + bdg[0];
    float D = (rd > 20.f) ? rd : log1pf(__expf(rd));   // softplus
    Dm[p] = (D + 1e-5f) * m;
#pragma unroll
    for (int r = 0; r < RANKK; ++r) Vm[(size_t)p * RANKK + r] = (av[r] + blr[r]) * m;
  }
}

// =====================================================================
// Kernel 4: per-batch Cov build + Cholesky + forward solve + stats.
// One block per batch, 128x128 tile in dynamic LDS.
// =====================================================================
__device__ __forceinline__ float block_reduce(float v, float* red) {
  __syncthreads();
  int tid = threadIdx.x;
#pragma unroll
  for (int s = 16; s; s >>= 1) v += __shfl_xor(v, s, 32);
  if ((tid & 31) == 0) red[tid >> 5] = v;
  __syncthreads();
  if (tid == 0) {
    float s = 0.f;
    for (int i = 0; i < 8; ++i) s += red[i];
    red[0] = s;
  }
  __syncthreads();
  return red[0];
}

__global__ __launch_bounds__(256) void k_chol(
    const float* __restrict__ Vm, const float* __restrict__ Dm,
    const float* __restrict__ mean_out, const float* __restrict__ labels,
    const int* __restrict__ lengths, float* __restrict__ Cov_out,
    float* __restrict__ logp, float* __restrict__ smean,
    float* __restrict__ aerr, float* __restrict__ vsum,
    float* __restrict__ pm_out, float* __restrict__ lo_out,
    float* __restrict__ up_out) {
  extern __shared__ char smem[];
  float* A  = (float*)smem;        // 128*128
  float* Vl = A + NN * NN;         // 128*16
  float* Dl = Vl + NN * RANKK;     // 128
  float* yv = Dl + NN;             // 128
  __shared__ float red[8];

  const int b = blockIdx.x, tid = threadIdx.x;
  const int len = lengths[b];

  for (int i = tid; i < NN * RANKK; i += 256) Vl[i] = Vm[(size_t)b * NN * RANKK + i];
  for (int i = tid; i < NN; i += 256) Dl[i] = Dm[(size_t)b * NN + i];
  __syncthreads();

  // build Cov (also accumulate total sum => var_sum) and stream to d_out
  float covsum = 0.f;
  for (int i = tid; i < NN * NN; i += 256) {
    int r = i >> 7, c = i & 127;
    float acc = 0.f;
#pragma unroll
    for (int k = 0; k < RANKK; ++k) acc += Vl[r * RANKK + k] * Vl[c * RANKK + k];
    if (r == c) acc += Dl[r];
    A[i] = acc;
    covsum += acc;
    Cov_out[(size_t)b * NN * NN + i] = acc;
  }
  __syncthreads();
  // identity padding on invalid diag
  for (int i = tid; i < NN; i += 256)
    if (i >= len) A[i * NN + i] += 1.f;
  __syncthreads();

  // right-looking Cholesky (lower)
  for (int k = 0; k < NN; ++k) {
    if (tid == 0) A[k * NN + k] = sqrtf(A[k * NN + k]);
    __syncthreads();
    float dk = A[k * NN + k];
    for (int i = k + 1 + tid; i < NN; i += 256) A[i * NN + k] /= dk;
    __syncthreads();
    int rem = NN - 1 - k;
    for (int idx = tid; idx < rem * rem; idx += 256) {
      int r = idx / rem, c = idx % rem;
      int i = k + 1 + r, j = k + 1 + c;
      if (j <= i) A[i * NN + j] -= A[i * NN + k] * A[j * NN + k];
    }
    __syncthreads();
  }

  // forward solve L y = (labels - mean) * mask
  for (int i = tid; i < NN; i += 256) {
    float m = (i < len) ? 1.f : 0.f;
    yv[i] = (labels[(size_t)b * NN + i] - mean_out[(size_t)b * NN + i]) * m;
  }
  __syncthreads();
  for (int k = 0; k < NN; ++k) {
    if (tid == 0) yv[k] /= A[k * NN + k];
    __syncthreads();
    float yk = yv[k];
    for (int i = k + 1 + tid; i < NN; i += 256) yv[i] -= A[i * NN + k] * yk;
    __syncthreads();
  }

  // per-batch stats
  float q = 0.f, ld = 0.f, sm = 0.f, ae = 0.f;
  for (int i = tid; i < NN; i += 256) {
    float m = (i < len) ? 1.f : 0.f;
    q  += yv[i] * yv[i];
    ld += 2.f * logf(A[i * NN + i]) * m;
    float me = mean_out[(size_t)b * NN + i];
    sm += me * m;
    ae += fabsf(me - labels[(size_t)b * NN + i]) * m;
  }
  q  = block_reduce(q, red);
  ld = block_reduce(ld, red);
  sm = block_reduce(sm, red);
  ae = block_reduce(ae, red);
  covsum = block_reduce(covsum, red);

  if (tid == 0) {
    const float LOG2PI = 1.8378770664093453f;
    logp[b]  = -0.5f * (q + ld + (float)len * LOG2PI);
    smean[b] = sm;
    aerr[b]  = ae;
    vsum[b]  = covsum;
    pm_out[b] = sm;
    const float z = 1.9599639845400545f;   // ndtri(0.975)
    float ss = sqrtf(covsum);
    lo_out[b] = sm - z * ss;
    up_out[b] = sm + z * ss;
  }
}

// =====================================================================
// Kernel 5: final scalar loss
// =====================================================================
__global__ __launch_bounds__(256) void k_final(
    const float* __restrict__ logp, const float* __restrict__ aerr,
    const float* __restrict__ smean, const int* __restrict__ lengths,
    const float* __restrict__ td, float* __restrict__ loss_out) {
  __shared__ float red[8];
  int tid = threadIdx.x;
  float acc[5] = {0.f, 0.f, 0.f, 0.f, 0.f};
  for (int b = tid; b < BB; b += 256) {
    acc[0] += logp[b];
    acc[1] += aerr[b];
    acc[2] += (float)lengths[b];
    float t = td[b];
    if (t != 0.f) { acc[3] += fabsf((smean[b] - t) / t); acc[4] += 1.f; }
  }
  float res[5];
#pragma unroll
  for (int v = 0; v < 5; ++v) {
    float x = acc[v];
#pragma unroll
    for (int s = 16; s; s >>= 1) x += __shfl_xor(x, s, 32);
    if ((tid & 31) == 0) red[tid >> 5] = x;
    __syncthreads();
    if (tid == 0) {
      float s = 0.f;
      for (int i = 0; i < 8; ++i) s += red[i];
      res[v] = s;
    }
    __syncthreads();
  }
  if (tid == 0) {
    float nll = -res[0] / (float)BB;
    float lm  = res[1] / res[2];
    float eta = res[3] / fmaxf(res[4], 1.f);
    loss_out[0] = eta + lm + nll;
  }
}

// =====================================================================
// host launcher
// =====================================================================
extern "C" void kernel_launch(void* const* d_in, const int* in_sizes, int n_in,
                              void* d_out, int out_size, void* d_ws, size_t ws_size,
                              hipStream_t stream) {
  (void)in_sizes; (void)n_in; (void)out_size; (void)ws_size;
  const int*   xs       = (const int*)d_in[0];
  const float* dist     = (const float*)d_in[1];
  const int*   lengths  = (const int*)d_in[2];
  const int*   start_ts = (const int*)d_in[3];
  const float* labels   = (const float*)d_in[4];
  const float* td       = (const float*)d_in[5];
  const float* seg_emb  = (const float*)d_in[6];
  const float* slice_emb= (const float*)d_in[7];
  const float* Wd   = (const float*)d_in[8];
  const float* bd   = (const float*)d_in[9];
  const float* W1   = (const float*)d_in[10];
  const float* b1   = (const float*)d_in[11];
  const float* W2   = (const float*)d_in[12];
  const float* b2   = (const float*)d_in[13];
  const float* W_ih = (const float*)d_in[14];
  const float* W_hh = (const float*)d_in[15];
  const float* b_ih = (const float*)d_in[16];
  const float* b_hh = (const float*)d_in[17];
  const float* Wm   = (const float*)d_in[18];
  const float* bm   = (const float*)d_in[19];
  const float* Wdg  = (const float*)d_in[20];
  const float* bdg  = (const float*)d_in[21];
  const float* Wlr  = (const float*)d_in[22];
  const float* blr  = (const float*)d_in[23];

  // workspace carve-up
  char* ws = (char*)d_ws;
  size_t off = 0;
  auto carve = [&](size_t bytes) -> char* {
    char* p = ws + off;
    off = (off + bytes + 255) & ~(size_t)255;
    return p;
  };
  __bf16* W1t  = (__bf16*)carve((size_t)HH * IN1 * 2);
  __bf16* W2t  = (__bf16*)carve((size_t)HH * HH * 2);
  __bf16* Wihb = (__bf16*)carve((size_t)G3H * HH * 2);
  __bf16* Whhb = (__bf16*)carve((size_t)G3H * HH * 2);
  __bf16* xbuf = (__bf16*)carve((size_t)BB * NN * HH * 2);
  float*  outb = (float*)carve((size_t)BB * NN * HH * 4);
  float*  Vmb  = (float*)carve((size_t)BB * NN * RANKK * 4);
  float*  Dmb  = (float*)carve((size_t)BB * NN * 4);
  float*  logp = (float*)carve((size_t)BB * 4);
  float*  smn  = (float*)carve((size_t)BB * 4);
  float*  aer  = (float*)carve((size_t)BB * 4);
  float*  vsm  = (float*)carve((size_t)BB * 4);

  float* out_f   = (float*)d_out;
  float* loss_p  = out_f + OFF_LOSS;
  float* pm_p    = out_f + OFF_PM;
  float* mean_p  = out_f + OFF_MEAN;
  float* cov_p   = out_f + OFF_COV;
  float* lo_p    = out_f + OFF_LOWER;
  float* up_p    = out_f + OFF_UPPER;

  const int GRU_SMEM  = 3 * 32 * HH * 2 + 32 * G3H * 4 * 2;   // 245760 B
  const int CHOL_SMEM = (NN * NN + NN * RANKK + NN + NN) * 4; // 74752 B
  (void)hipFuncSetAttribute(reinterpret_cast<const void*>(k_gru),
                            hipFuncAttributeMaxDynamicSharedMemorySize, GRU_SMEM);
  (void)hipFuncSetAttribute(reinterpret_cast<const void*>(k_chol),
                            hipFuncAttributeMaxDynamicSharedMemorySize, CHOL_SMEM);

  k_prep<<<(G3H * HH + 255) / 256, 256, 0, stream>>>(W1, W2, W_ih, W_hh,
                                                     W1t, W2t, Wihb, Whhb);
  k_featmlp<<<(BB * NN) / 32, 256, 0, stream>>>(xs, dist, start_ts, seg_emb,
                                                slice_emb, Wd, bd, W1t, b1,
                                                W2t, b2, xbuf);
  k_gru<<<BB / 32, 256, GRU_SMEM, stream>>>(xbuf, Wihb, Whhb, b_ih, b_hh,
                                            lengths, outb);
  k_heads<<<(BB * NN) / 8, 256, 0, stream>>>(outb, Wm, bm, Wdg, bdg, Wlr, blr,
                                             lengths, mean_p, Vmb, Dmb);
  k_chol<<<BB, 256, CHOL_SMEM, stream>>>(Vmb, Dmb, mean_p, labels, lengths,
                                         cov_p, logp, smn, aer, vsm,
                                         pm_p, lo_p, up_p);
  k_final<<<1, 256, 0, stream>>>(logp, aer, smn, lengths, td, loss_p);
}
